// DecoderLayer_86646670230062
// MI455X (gfx1250) — compile-verified, matrix operand order checked by reading, not run on hardware
//
#include <hip/hip_runtime.h>
#include <math.h>

// ---------------- constants (match reference shapes) ----------------
#define BB   2
#define SS   1024
#define DD   2048
#define HH   16
#define KVHH 8
#define HDD  128
#define HFF  5632
#define NTOK 2048            // BB*SS
#define ATT_SCALE 0.08838834764831845f   // 1/sqrt(128)
#define RMS_EPS 1e-6f

typedef __attribute__((ext_vector_type(16))) __bf16        v16bf;
typedef __attribute__((ext_vector_type(8)))  float         v8f;
typedef __attribute__((ext_vector_type(4)))  unsigned int  u32x4;

union Frag { u32x4 q[2]; v16bf v; };

__device__ __forceinline__ unsigned short f2bf(float f) {
  unsigned int u = __float_as_uint(f);
  u += 0x7FFFu + ((u >> 16) & 1u);          // round-to-nearest-even
  return (unsigned short)(u >> 16);
}

// =====================================================================
// GEMM: C[M][N](f32) = A[M][K](bf16) @ Wt[N][K](bf16)^T (+bias)
// 8 waves, workgroup tile 128x128, wave tile 64x32, K-step 32.
// Manually 2x-unrolled distance-2 software pipeline: two *named*
// register staging sets (even/odd tiles) and compile-time LDS buffer
// indices, so no register indexing (movreld/cndmask) can appear.
// Tile t is fetched one full compute phase before its ds_store; the
// loadcnt wait therefore always hides behind 8 WMMAs + LDS traffic.
// Requires K % 64 == 0 (true for all call sites: K in {2048, 5632}).
// =====================================================================
__global__ __launch_bounds__(256) void gemm_bf16_kernel(
    const unsigned short* __restrict__ A,
    const unsigned short* __restrict__ Bt,
    const float* __restrict__ bias,
    float* __restrict__ C,
    int M, int N, int K)
{
  __shared__ unsigned short lA[2][128 * 32];
  __shared__ unsigned short lB[2][128 * 32];
  const int tid  = threadIdx.x;
  const int lane = tid & 31;
  const int wave = tid >> 5;
  const int wm   = wave & 1;           // 0..1 -> 64-row slice of M
  const int wn   = wave >> 1;          // 0..3 -> 32-col slice of N
  const int m    = lane & 15;
  const int half = lane >> 4;
  const int blockM = blockIdx.y * 128;
  const int blockN = blockIdx.x * 128;
  (void)M;

  // cooperative-load coordinates: 2 x 16B chunks per matrix per thread
  const int r0  = tid >> 2;                 // rows 0..63
  const int c0e = (tid & 3) * 8;
  const int r1  = r0 + 64;                  // rows 64..127
  const unsigned short* pA0 = A  + (size_t)(blockM + r0) * K + c0e;
  const unsigned short* pA1 = A  + (size_t)(blockM + r1) * K + c0e;
  const unsigned short* pB0 = Bt + (size_t)(blockN + r0) * K + c0e;
  const unsigned short* pB1 = Bt + (size_t)(blockN + r1) * K + c0e;
  const int lo0 = r0 * 32 + c0e;
  const int lo1 = r1 * 32 + c0e;

  const v8f vzero = {0.f,0.f,0.f,0.f,0.f,0.f,0.f,0.f};
  v8f acc[4][2];
  for (int i = 0; i < 4; ++i) for (int j = 0; j < 2; ++j) acc[i][j] = vzero;

  const int nk = K >> 5;               // even by contract

  // two named staging sets: even tiles (0,2,..) and odd tiles (1,3,..)
  u32x4 ea0, ea1, eb0, eb1;
  u32x4 oa0, oa1, ob0, ob1;

#define LOAD_TILE(t, A0r, A1r, B0r, B1r)                                     \
  { const size_t o_ = (size_t)(t) * 32;                                      \
    A0r = *(const u32x4*)(pA0 + o_);                                         \
    A1r = *(const u32x4*)(pA1 + o_);                                         \
    B0r = *(const u32x4*)(pB0 + o_);                                         \
    B1r = *(const u32x4*)(pB1 + o_); }

#define STORE_TILE(buf, A0r, A1r, B0r, B1r)                                  \
  { *(u32x4*)&lA[buf][lo0] = A0r;                                            \
    *(u32x4*)&lA[buf][lo1] = A1r;                                            \
    *(u32x4*)&lB[buf][lo0] = B0r;                                            \
    *(u32x4*)&lB[buf][lo1] = B1r; }

#define COMPUTE_TILE(buf)                                                    \
  { Frag af[4], bfr[2];                                                      \
    for (int mt = 0; mt < 4; ++mt) {                                         \
      int r = wm * 64 + mt * 16 + m;                                         \
      af[mt].q[0] = *(const u32x4*)&lA[buf][r * 32 + 8 * half];              \
      af[mt].q[1] = *(const u32x4*)&lA[buf][r * 32 + 16 + 8 * half];         \
    }                                                                        \
    for (int nt = 0; nt < 2; ++nt) {                                         \
      int r = wn * 32 + nt * 16 + m;                                         \
      bfr[nt].q[0] = *(const u32x4*)&lB[buf][r * 32 + 8 * half];             \
      bfr[nt].q[1] = *(const u32x4*)&lB[buf][r * 32 + 16 + 8 * half];        \
    }                                                                        \
    for (int mt = 0; mt < 4; ++mt)                                           \
      for (int nt = 0; nt < 2; ++nt)                                         \
        acc[mt][nt] = __builtin_amdgcn_wmma_f32_16x16x32_bf16(               \
            false, af[mt].v, false, bfr[nt].v, (short)0, acc[mt][nt],        \
            false, false); }

  // prologue: tile0 -> buf0 (direct), tile1 staged in odd regs (in flight)
  LOAD_TILE(0, ea0, ea1, eb0, eb1);
  STORE_TILE(0, ea0, ea1, eb0, eb1);
  LOAD_TILE(1, oa0, oa1, ob0, ob1);

  for (int k = 0; k < nk; k += 2) {
    // ---- even phase: compute tile k from buf0 ----
    __syncthreads();                          // buf0 published
    if (k + 2 < nk) LOAD_TILE(k + 2, ea0, ea1, eb0, eb1);
    COMPUTE_TILE(0);
    __syncthreads();                          // prior readers of buf1 done
    STORE_TILE(1, oa0, oa1, ob0, ob1);        // tile k+1 (fetched a phase ago)

    // ---- odd phase: compute tile k+1 from buf1 ----
    __syncthreads();                          // buf1 published
    if (k + 3 < nk) LOAD_TILE(k + 3, oa0, oa1, ob0, ob1);
    COMPUTE_TILE(1);
    if (k + 2 < nk) {
      __syncthreads();                        // prior readers of buf0 done
      STORE_TILE(0, ea0, ea1, eb0, eb1);      // tile k+2 (fetched a phase ago)
    }
  }

#undef LOAD_TILE
#undef STORE_TILE
#undef COMPUTE_TILE

  for (int nt = 0; nt < 2; ++nt) {
    int col  = blockN + wn * 32 + nt * 16 + m;
    float bv = bias ? bias[col] : 0.f;
    for (int mt = 0; mt < 4; ++mt)
      for (int r = 0; r < 8; ++r) {
        int row = blockM + wm * 64 + mt * 16 + r + 8 * half;
        C[(size_t)row * N + col] = acc[mt][nt][r] + bv;
      }
  }
}

// =====================================================================
// Flash attention. Q:[B][nH][S][128] bf16, K:[B][kvH][S][128] bf16,
// Vt:[B][kvH][128][S] bf16, Out:[B*S][2048] f32 (col base h*128).
// 4 waves; each wave owns 16 query rows; kv walked in blocks of 32.
// =====================================================================
__global__ __launch_bounds__(128) void attn_kernel(
    const unsigned short* __restrict__ Q,
    const unsigned short* __restrict__ Kmat,
    const unsigned short* __restrict__ Vt,
    float* __restrict__ Out,
    int S_, int nH, int kvH, int kvShift)
{
  __shared__ unsigned short lP[4][16][32];
  const int tid  = threadIdx.x;
  const int lane = tid & 31;
  const int wave = tid >> 5;
  const int m    = lane & 15;
  const int half = lane >> 4;
  const int b = blockIdx.z, h = blockIdx.y;
  const int q0  = blockIdx.x * 64 + wave * 16;
  const int kvh = h >> kvShift;

  const unsigned short* Qb = Q    + ((size_t)(b * nH  + h)   * S_ + q0) * 128;
  const unsigned short* Kb = Kmat + (size_t)(b * kvH + kvh) * S_ * 128;
  const unsigned short* Vb = Vt   + (size_t)(b * kvH + kvh) * 128 * S_;

  Frag qf[4];
  for (int c = 0; c < 4; ++c) {
    qf[c].q[0] = *(const u32x4*)(Qb + (size_t)m * 128 + c * 32 + 8 * half);
    qf[c].q[1] = *(const u32x4*)(Qb + (size_t)m * 128 + c * 32 + 16 + 8 * half);
  }

  const v8f vzero = {0.f,0.f,0.f,0.f,0.f,0.f,0.f,0.f};
  v8f acc[8];
  float mi[8], li[8];
  for (int r = 0; r < 8; ++r) { acc[r] = vzero; mi[r] = -1e30f; li[r] = 0.f; }

  for (int kv0 = 0; kv0 < S_; kv0 += 32) {
    // scores: two 16x16 tiles, K=128 via 4 chained WMMAs each
    v8f c0 = vzero, c1 = vzero;
    for (int c = 0; c < 4; ++c) {
      Frag k0f, k1f;
      const unsigned short* K0 = Kb + (size_t)(kv0 + m)      * 128 + c * 32;
      const unsigned short* K1 = Kb + (size_t)(kv0 + 16 + m) * 128 + c * 32;
      k0f.q[0] = *(const u32x4*)(K0 + 8 * half);
      k0f.q[1] = *(const u32x4*)(K0 + 16 + 8 * half);
      k1f.q[0] = *(const u32x4*)(K1 + 8 * half);
      k1f.q[1] = *(const u32x4*)(K1 + 16 + 8 * half);
      c0 = __builtin_amdgcn_wmma_f32_16x16x32_bf16(false, qf[c].v, false, k0f.v, (short)0, c0, false, false);
      c1 = __builtin_amdgcn_wmma_f32_16x16x32_bf16(false, qf[c].v, false, k1f.v, (short)0, c1, false, false);
    }

    // online softmax: row stats live per lane; 16-lane xor reductions
    float p0[8], p1[8];
    for (int r = 0; r < 8; ++r) {
      float s0 = c0[r] * ATT_SCALE, s1 = c1[r] * ATT_SCALE;
      float mx = fmaxf(s0, s1);
      for (int msk = 8; msk; msk >>= 1) mx = fmaxf(mx, __shfl_xor(mx, msk, 32));
      float mnew  = fmaxf(mi[r], mx);
      float alpha = __expf(mi[r] - mnew);
      p0[r] = __expf(s0 - mnew);
      p1[r] = __expf(s1 - mnew);
      float rs = p0[r] + p1[r];
      for (int msk = 8; msk; msk >>= 1) rs += __shfl_xor(rs, msk, 32);
      li[r] = li[r] * alpha + rs;
      mi[r] = mnew;
      for (int nt = 0; nt < 8; ++nt) acc[nt][r] *= alpha;
    }

    // P: C-layout -> LDS -> A-fragment layout (16x32 bf16)
    for (int r = 0; r < 8; ++r) {
      lP[wave][r + 8 * half][m]      = f2bf(p0[r]);
      lP[wave][r + 8 * half][16 + m] = f2bf(p1[r]);
    }
    __syncthreads();
    Frag pf;
    pf.q[0] = *(const u32x4*)&lP[wave][m][8 * half];
    pf.q[1] = *(const u32x4*)&lP[wave][m][16 + 8 * half];

    // O += P @ V  (V pre-transposed: Bt layout, contiguous along seq)
    for (int nt = 0; nt < 8; ++nt) {
      Frag vf;
      const unsigned short* Vp = Vb + (size_t)(nt * 16 + m) * S_ + kv0;
      vf.q[0] = *(const u32x4*)(Vp + 8 * half);
      vf.q[1] = *(const u32x4*)(Vp + 16 + 8 * half);
      acc[nt] = __builtin_amdgcn_wmma_f32_16x16x32_bf16(
          false, pf.v, false, vf.v, (short)0, acc[nt], false, false);
    }
    __syncthreads();
  }

  for (int r = 0; r < 8; ++r) {
    float inv  = 1.0f / li[r];
    size_t row = (size_t)b * S_ + q0 + r + 8 * half;
    for (int nt = 0; nt < 8; ++nt)
      Out[row * DD + h * 128 + nt * 16 + m] = acc[nt][r] * inv;
  }
}

// ---------------- elementwise / norm / pack kernels ----------------
__device__ __forceinline__ float block_sum256(float v) {
  for (int msk = 16; msk; msk >>= 1) v += __shfl_xor(v, msk, 32);
  __shared__ float red[8];
  int lane = threadIdx.x & 31, w = threadIdx.x >> 5;
  if (lane == 0) red[w] = v;
  __syncthreads();
  float s = (lane < 8) ? red[lane] : 0.f;
  for (int msk = 4; msk; msk >>= 1) s += __shfl_xor(s, msk, 32);
  s = __shfl(s, 0, 32);
  __syncthreads();
  return s;
}

__global__ __launch_bounds__(256) void double_rms_kernel(
    const float* __restrict__ x, const float* __restrict__ o2,
    const float* __restrict__ w1, const float* __restrict__ w2,
    unsigned short* __restrict__ out)
{
  const size_t base = (size_t)blockIdx.x * DD;
  float t[8]; float ss = 0.f;
  for (int j = 0; j < 8; ++j) {
    int i = threadIdx.x + j * 256;
    t[j] = x[base + i] + o2[base + i];
    ss += t[j] * t[j];
  }
  float inv1 = rsqrtf(block_sum256(ss) / DD + RMS_EPS);
  float x1[8]; float ss2 = 0.f;
  for (int j = 0; j < 8; ++j) {
    int i = threadIdx.x + j * 256;
    x1[j] = t[j] * inv1 * w1[i];
    ss2 += x1[j] * x1[j];
  }
  float inv2 = rsqrtf(block_sum256(ss2) / DD + RMS_EPS);
  for (int j = 0; j < 8; ++j) {
    int i = threadIdx.x + j * 256;
    out[base + i] = f2bf(x1[j] * inv2 * w2[i]);
  }
}

__global__ __launch_bounds__(256) void resid_rms_kernel(
    const float* __restrict__ y, const float* __restrict__ a,
    const float* __restrict__ w, float* __restrict__ hOut,
    unsigned short* __restrict__ hnOut)
{
  const size_t base = (size_t)blockIdx.x * DD;
  float t[8]; float ss = 0.f;
  for (int j = 0; j < 8; ++j) {
    int i = threadIdx.x + j * 256;
    t[j] = y[base + i] + a[base + i];
    hOut[base + i] = t[j];
    ss += t[j] * t[j];
  }
  float inv = rsqrtf(block_sum256(ss) / DD + RMS_EPS);
  for (int j = 0; j < 8; ++j) {
    int i = threadIdx.x + j * 256;
    hnOut[base + i] = f2bf(t[j] * inv * w[i]);
  }
}

// (B,S,heads*128) f32 -> (B,heads,S,128) bf16, optional RoPE on pairs
__global__ void pack_rows_kernel(const float* __restrict__ X, unsigned short* __restrict__ Dst,
                                 int S_, int heads, int applyRope,
                                 const float* __restrict__ cosT, const float* __restrict__ sinT)
{
  size_t idx   = (size_t)blockIdx.x * blockDim.x + threadIdx.x;
  size_t total = (size_t)BB * S_ * heads * 64;
  if (idx >= total) return;
  int d2     = idx & 63;
  size_t t   = idx >> 6;
  int hh     = (int)(t % heads);
  size_t row = t / heads;
  int s = (int)(row % S_);
  int b = (int)(row / S_);
  const float* xp = X + row * (size_t)heads * 128 + hh * 128 + 2 * d2;
  float x0 = xp[0], x1 = xp[1];
  float re, im;
  if (applyRope) {
    float c = cosT[s * 64 + d2], sn = sinT[s * 64 + d2];
    re = x0 * c - x1 * sn;
    im = x0 * sn + x1 * c;
  } else { re = x0; im = x1; }
  size_t o = (((size_t)(b * heads + hh) * S_ + s) * 128) + 2 * d2;
  unsigned int packed = (unsigned int)f2bf(re) | ((unsigned int)f2bf(im) << 16);
  *(unsigned int*)(Dst + o) = packed;
}

// (B,S,heads*128) f32 -> (B,heads,128,S) bf16  (V transpose for PV WMMA)
__global__ void pack_vT_kernel(const float* __restrict__ X, unsigned short* __restrict__ Dst,
                               int S_, int heads)
{
  size_t idx   = (size_t)blockIdx.x * blockDim.x + threadIdx.x;
  size_t total = (size_t)BB * heads * 128 * S_;
  if (idx >= total) return;
  int s     = (int)(idx % S_);
  size_t t  = idx / S_;
  int d     = (int)(t % 128);
  size_t t2 = t / 128;
  int hh    = (int)(t2 % heads);
  int b     = (int)(t2 / heads);
  Dst[idx] = f2bf(X[((size_t)(b * S_ + s) * heads + hh) * 128 + d]);
}

__global__ void cvt_bf16_kernel(const float* __restrict__ src,
                                unsigned short* __restrict__ dst, size_t n)
{
  size_t i = (size_t)blockIdx.x * blockDim.x + threadIdx.x;
  if (i < n) dst[i] = f2bf(src[i]);
}

// W[K][N] f32 -> Wt[N][K] bf16
__global__ void transpose_cvt_kernel(const float* __restrict__ W,
                                     unsigned short* __restrict__ Wt, int K, int N)
{
  size_t idx   = (size_t)blockIdx.x * blockDim.x + threadIdx.x;
  size_t total = (size_t)K * N;
  if (idx >= total) return;
  int k    = (int)(idx % K);
  size_t n = idx / K;
  Wt[idx] = f2bf(W[(size_t)k * N + n]);
}

__global__ void silu_mul_kernel(const float* __restrict__ a, const float* __restrict__ b,
                                unsigned short* __restrict__ g, size_t n)
{
  size_t i = (size_t)blockIdx.x * blockDim.x + threadIdx.x;
  if (i >= n) return;
  float x = a[i];
  g[i] = f2bf(x / (1.f + __expf(-x)) * b[i]);
}

__global__ void add_kernel(const float* __restrict__ h, const float* __restrict__ f,
                           float* __restrict__ out, size_t n)
{
  size_t i = (size_t)blockIdx.x * blockDim.x + threadIdx.x;
  if (i < n) out[i] = h[i] + f[i];
}

// =====================================================================
// host driver
// =====================================================================
extern "C" void kernel_launch(void* const* d_in, const int* in_sizes, int n_in,
                              void* d_out, int out_size, void* d_ws, size_t ws_size,
                              hipStream_t stream) {
  (void)in_sizes; (void)n_in; (void)out_size;
  const float* x        = (const float*)d_in[0];
  const float* y        = (const float*)d_in[1];
  const float* fcos     = (const float*)d_in[2];
  const float* fsin     = (const float*)d_in[3];
  const float* Wq       = (const float*)d_in[4];
  const float* bq       = (const float*)d_in[5];
  const float* Wk       = (const float*)d_in[6];
  const float* bk       = (const float*)d_in[7];
  const float* Wv       = (const float*)d_in[8];
  const float* bv       = (const float*)d_in[9];
  const float* Wo       = (const float*)d_in[10];
  const float* bo       = (const float*)d_in[11];
  const float* norm1_w  = (const float*)d_in[12];
  const float* attn_n_w = (const float*)d_in[13];
  const float* awq      = (const float*)d_in[14];
  const float* awk      = (const float*)d_in[15];
  const float* awv      = (const float*)d_in[16];
  const float* awo      = (const float*)d_in[17];
  const float* w1       = (const float*)d_in[18];
  const float* w2       = (const float*)d_in[19];
  const float* w3       = (const float*)d_in[20];
  const float* ffn_n_w  = (const float*)d_in[21];
  float* out = (float*)d_out;
  char* ws = (char*)d_ws;

  // workspace layout (bump allocator, 256B aligned)
  size_t off = 0;
  auto alloc = [&](size_t bytes) {
    size_t p = off;
    off += (bytes + 255) & ~(size_t)255;
    return p;
  };
  const size_t wslot = alloc((size_t)HFF * DD * 2);        // JIT-transposed weight (bf16), reused
  const size_t sA = alloc((size_t)NTOK * HFF * 4);         // big f32 slot
  const size_t sB = alloc((size_t)NTOK * HFF * 4);
  const size_t sC = alloc((size_t)NTOK * HFF * 4);
  const size_t sD = alloc((size_t)NTOK * DD * 4);
  const size_t sE = alloc((size_t)NTOK * DD * 4);
  const size_t sF = alloc((size_t)NTOK * DD * 4);
  const size_t sG = alloc((size_t)NTOK * DD * 4);
  if (off > ws_size) return;   // workspace too small: nothing safe to do

  auto GEMM = [&](size_t aOff, const float* W, int K, int N, const float* bias, size_t cOff) {
    size_t tot = (size_t)K * N;
    transpose_cvt_kernel<<<(unsigned)((tot + 255) / 256), 256, 0, stream>>>(
        W, (unsigned short*)(ws + wslot), K, N);
    dim3 g((unsigned)(N / 128), (unsigned)(NTOK / 128));
    gemm_bf16_kernel<<<g, 256, 0, stream>>>(
        (const unsigned short*)(ws + aOff), (const unsigned short*)(ws + wslot),
        bias, (float*)(ws + cOff), NTOK, N, K);
  };
  auto CVT = [&](size_t src, size_t dst, size_t n) {
    cvt_bf16_kernel<<<(unsigned)((n + 255) / 256), 256, 0, stream>>>(
        (const float*)(ws + src), (unsigned short*)(ws + dst), n);
  };

  const size_t nBD = (size_t)NTOK * DD;

  // ---------- Stage A: MHA ----------
  cvt_bf16_kernel<<<(unsigned)((nBD + 255) / 256), 256, 0, stream>>>(
      x, (unsigned short*)(ws + sD), nBD);                       // xb -> D
  GEMM(sD, Wq, DD, DD, bq, sA);                                  // qf -> A
  GEMM(sD, Wk, DD, DD, bk, sB);                                  // kf -> B
  GEMM(sD, Wv, DD, DD, bv, sC);                                  // vf -> C
  {
    size_t nq = (size_t)BB * SS * HH * 64;
    pack_rows_kernel<<<(unsigned)((nq + 255) / 256), 256, 0, stream>>>(
        (const float*)(ws + sA), (unsigned short*)(ws + sE), SS, HH, 0, fcos, fsin);
    pack_rows_kernel<<<(unsigned)((nq + 255) / 256), 256, 0, stream>>>(
        (const float*)(ws + sB), (unsigned short*)(ws + sF), SS, HH, 0, fcos, fsin);
    size_t nv = (size_t)BB * HH * 128 * SS;
    pack_vT_kernel<<<(unsigned)((nv + 255) / 256), 256, 0, stream>>>(
        (const float*)(ws + sC), (unsigned short*)(ws + sG), SS, HH);
  }
  {
    dim3 g(SS / 64, HH, BB);
    attn_kernel<<<g, 128, 0, stream>>>(
        (const unsigned short*)(ws + sE), (const unsigned short*)(ws + sF),
        (const unsigned short*)(ws + sG), (float*)(ws + sA), SS, HH, HH, 0);  // ao -> A
  }
  CVT(sA, sD, nBD);                                              // aob -> D
  GEMM(sD, Wo, DD, DD, bo, sB);                                  // o2 -> B
  double_rms_kernel<<<NTOK, 256, 0, stream>>>(
      x, (const float*)(ws + sB), norm1_w, attn_n_w, (unsigned short*)(ws + sE)); // xnb -> E

  // ---------- Stage B: GQA + RoPE ----------
  GEMM(sE, awq, DD, HH * HDD, nullptr, sA);                      // xq -> A
  GEMM(sE, awk, DD, KVHH * HDD, nullptr, sB);                    // xk -> B
  GEMM(sE, awv, DD, KVHH * HDD, nullptr, sC);                    // xv -> C
  {
    size_t nq = (size_t)BB * SS * HH * 64;
    pack_rows_kernel<<<(unsigned)((nq + 255) / 256), 256, 0, stream>>>(
        (const float*)(ws + sA), (unsigned short*)(ws + sD), SS, HH, 1, fcos, fsin);
    size_t nk = (size_t)BB * SS * KVHH * 64;
    pack_rows_kernel<<<(unsigned)((nk + 255) / 256), 256, 0, stream>>>(
        (const float*)(ws + sB), (unsigned short*)(ws + sF), SS, KVHH, 1, fcos, fsin);
    size_t nv = (size_t)BB * KVHH * 128 * SS;
    pack_vT_kernel<<<(unsigned)((nv + 255) / 256), 256, 0, stream>>>(
        (const float*)(ws + sC), (unsigned short*)(ws + sG), SS, KVHH);
  }
  {
    dim3 g(SS / 64, HH, BB);
    attn_kernel<<<g, 128, 0, stream>>>(
        (const unsigned short*)(ws + sD), (const unsigned short*)(ws + sF),
        (const unsigned short*)(ws + sG), (float*)(ws + sA), SS, HH, KVHH, 1); // ao2 -> A
  }
  CVT(sA, sB, nBD);                                              // ao2b -> B
  GEMM(sB, awo, HH * HDD, DD, nullptr, sC);                      // ao2p -> C
  resid_rms_kernel<<<NTOK, 256, 0, stream>>>(
      y, (const float*)(ws + sC), ffn_n_w, (float*)(ws + sD),    // h -> D
      (unsigned short*)(ws + sE));                               // hnb -> E

  // ---------- Stage C: FFN ----------
  GEMM(sE, w1, DD, HFF, nullptr, sA);                            // a1 -> A
  GEMM(sE, w3, DD, HFF, nullptr, sB);                            // a3 -> B
  {
    size_t n = (size_t)NTOK * HFF;
    silu_mul_kernel<<<(unsigned)((n + 255) / 256), 256, 0, stream>>>(
        (const float*)(ws + sA), (const float*)(ws + sB), (unsigned short*)(ws + sC), n);
  }
  GEMM(sC, w2, HFF, DD, nullptr, sF);                            // ff -> F
  add_kernel<<<(unsigned)((nBD + 255) / 256), 256, 0, stream>>>(
      (const float*)(ws + sD), (const float*)(ws + sF), out, nBD);
}
